// BiGru_67405216743861
// MI455X (gfx1250) — compile-verified, matrix operand order checked by reading, not run on hardware
//
#include <hip/hip_runtime.h>

// ---------------- problem constants ----------------
constexpr int Bc = 64, Tc = 512, Dc = 512, Hc = 1024, Oc = 512;
constexpr int BHc = Bc * Hc;                  // 65536
constexpr long OUT_MAIN = (long)Bc * Tc * Oc; // 16,777,216 floats

typedef __attribute__((ext_vector_type(16))) __bf16          bf16x16;
typedef __attribute__((ext_vector_type(8)))  float           f32x8;
typedef __attribute__((ext_vector_type(8)))  unsigned short  u16x8;
typedef __attribute__((ext_vector_type(16))) unsigned short  u16x16;

__device__ __forceinline__ unsigned short f2bf(float f) {
  union { float f; unsigned u; } v; v.f = f;
  unsigned u = v.u;
  u += 0x7FFFu + ((u >> 16) & 1u);   // round-to-nearest-even
  return (unsigned short)(u >> 16);
}

// 16x32 bf16 fragment load (A or B^T), rows = lane&15 relative to p,
// K layout per CDNA5 ISA: lanes 0-15 -> K{0..7,16..23}, lanes 16-31 -> K{8..15,24..31}
__device__ __forceinline__ bf16x16 load_frag(const unsigned short* __restrict__ p,
                                             long rstride, int lane) {
  const int r = lane & 15;
  const int half = lane >> 4;
  const unsigned short* q = p + (long)r * rstride + half * 8;
  u16x8 lo = *reinterpret_cast<const u16x8*>(q);
  u16x8 hi = *reinterpret_cast<const u16x8*>(q + 16);
  u16x16 u;
#pragma unroll
  for (int i = 0; i < 8; ++i) { u[i] = lo[i]; u[8 + i] = hi[i]; }
  return __builtin_bit_cast(bf16x16, u);
}

__device__ __forceinline__ f32x8 wmma_bf16(bf16x16 a, bf16x16 b, f32x8 c) {
  return __builtin_amdgcn_wmma_f32_16x16x32_bf16(false, a, false, b, (short)0, c,
                                                 false, false);
}

__device__ __forceinline__ float sigmoidf_(float x) { return 1.0f / (1.0f + __expf(-x)); }

// ---------------- preprocessing kernels ----------------
__global__ void cvt_bf16_kernel(const float* __restrict__ in,
                                unsigned short* __restrict__ out, long n) {
  long i = (long)blockIdx.x * blockDim.x + threadIdx.x;
  long stride = (long)gridDim.x * blockDim.x;
  for (; i < n; i += stride) out[i] = f2bf(in[i]);
}

// WT[n*Ktot + k] = (k<din ? Wx[k,n] : Wh[k-din,n]) as bf16 ; n in [0,3H)
__global__ void build_wt_kernel(const float* __restrict__ Wx,
                                const float* __restrict__ Wh,
                                unsigned short* __restrict__ WT,
                                int din, int Ktot) {
  long total = (long)(3 * Hc) * Ktot;
  long i = (long)blockIdx.x * blockDim.x + threadIdx.x;
  long stride = (long)gridDim.x * blockDim.x;
  for (; i < total; i += stride) {
    int n = (int)(i / Ktot);
    int k = (int)(i % Ktot);
    float v = (k < din) ? Wx[(long)k * (3 * Hc) + n]
                        : Wh[(long)(k - din) * (3 * Hc) + n];
    WT[i] = f2bf(v);
  }
}

// bias[0:H)=bx_r+bh_r, [H:2H)=bx_z+bh_z, [2H:3H)=bx_n, [3H:4H)=bh_n
__global__ void build_bias_kernel(const float* __restrict__ bx,
                                  const float* __restrict__ bh,
                                  float* __restrict__ bias) {
  int i = blockIdx.x * blockDim.x + threadIdx.x;
  if (i >= 4 * Hc) return;
  int j = i & (Hc - 1);
  int sec = i >> 10;
  float v;
  if (sec == 0)      v = bx[j] + bh[j];
  else if (sec == 1) v = bx[Hc + j] + bh[Hc + j];
  else if (sec == 2) v = bx[2 * Hc + j];
  else               v = bh[2 * Hc + j];
  bias[i] = v;
}

// fcWT[n*2048 + k] = fcW[k*512 + n]
__global__ void build_fcwt_kernel(const float* __restrict__ W,
                                  unsigned short* __restrict__ WT) {
  long total = (long)Oc * (2 * Hc);
  long i = (long)blockIdx.x * blockDim.x + threadIdx.x;
  long stride = (long)gridDim.x * blockDim.x;
  for (; i < total; i += stride) {
    int n = (int)(i / (2 * Hc));
    int k = (int)(i % (2 * Hc));
    WT[i] = f2bf(W[(long)k * Oc + n]);
  }
}

// zero h buffers (buf 0) for both directions
__global__ void zero_h_kernel(float* __restrict__ h32,
                              unsigned short* __restrict__ hbf) {
  int i = blockIdx.x * blockDim.x + threadIdx.x;
  if (i >= 2 * BHc) return;
  int d = i / BHc, r = i % BHc;
  h32[(size_t)d * 2 * BHc + r] = 0.0f;
  hbf[(size_t)d * 2 * BHc + r] = 0;
}

// ---------------- fused GRU step: gh+gx GEMM + gates ----------------
// grid (H/32, 2 dirs), block 256 (8 waves: wave = (mt in 0..3) x (nt in 0..1))
__global__ __launch_bounds__(256) void gru_step_kernel(
    const unsigned short* __restrict__ xseq_fw,
    const unsigned short* __restrict__ xseq_bw,
    int din, int Ktot,
    const unsigned short* __restrict__ WT_fw,
    const unsigned short* __restrict__ WT_bw,
    const float* __restrict__ bias_fw,
    const float* __restrict__ bias_bw,
    float* __restrict__ h32,
    unsigned short* __restrict__ hbf,
    unsigned short* __restrict__ y_fw,
    unsigned short* __restrict__ y_bw,
    int iter) {
  const int dir = blockIdx.y;
  const int t = dir ? (Tc - 1 - iter) : iter;
  const int rbuf = iter & 1, wbuf = rbuf ^ 1;

  const unsigned short* xseq = dir ? xseq_bw : xseq_fw;
  const unsigned short* WT   = dir ? WT_bw   : WT_fw;
  const float*          bias = dir ? bias_bw : bias_fw;
  unsigned short*       y    = dir ? y_bw    : y_fw;

  const float*          h_rd  = h32 + (size_t)dir * 2 * BHc + (size_t)rbuf * BHc;
  float*                h_wr  = h32 + (size_t)dir * 2 * BHc + (size_t)wbuf * BHc;
  const unsigned short* hb_rd = hbf + (size_t)dir * 2 * BHc + (size_t)rbuf * BHc;
  unsigned short*       hb_wr = hbf + (size_t)dir * 2 * BHc + (size_t)wbuf * BHc;

  const int tx = threadIdx.x, lane = tx & 31, wave = tx >> 5;
  const int mt = wave & 3, nt = wave >> 2;
  const int jbase = blockIdx.x * 32 + nt * 16;   // hidden-column base of this wave

  const long xr_stride = (long)Tc * din;
  const unsigned short* pa_x = xseq + (long)t * din + (long)(mt * 16) * xr_stride;
  const unsigned short* pa_h = hb_rd + (long)(mt * 16) * Hc;
  const unsigned short* pb_r = WT + (long)(0 * Hc + jbase) * Ktot;
  const unsigned short* pb_z = WT + (long)(1 * Hc + jbase) * Ktot;
  const unsigned short* pb_n = WT + (long)(2 * Hc + jbase) * Ktot;

  f32x8 cr{}, cz{}, cxn{}, chn{};

  // ---- x region: accumulate r, z, xn ----
#pragma unroll 2
  for (int kc = 0; kc < din; kc += 32) {
    bf16x16 a  = load_frag(pa_x + kc, xr_stride, lane);
    bf16x16 br = load_frag(pb_r + kc, Ktot, lane);
    bf16x16 bz = load_frag(pb_z + kc, Ktot, lane);
    bf16x16 bn = load_frag(pb_n + kc, Ktot, lane);
    __builtin_prefetch(pb_r + kc + 64, 0, 1);
    cr  = wmma_bf16(a, br, cr);
    cz  = wmma_bf16(a, bz, cz);
    cxn = wmma_bf16(a, bn, cxn);
  }

  // ---- h region: accumulate r, z, hn ----
#pragma unroll 2
  for (int kc = 0; kc < Hc; kc += 32) {
    bf16x16 a  = load_frag(pa_h + kc, Hc, lane);
    bf16x16 br = load_frag(pb_r + din + kc, Ktot, lane);
    bf16x16 bz = load_frag(pb_z + din + kc, Ktot, lane);
    bf16x16 bn = load_frag(pb_n + din + kc, Ktot, lane);
    __builtin_prefetch(pb_n + din + kc + 64, 0, 1);
    cr  = wmma_bf16(a, br, cr);
    cz  = wmma_bf16(a, bz, cz);
    chn = wmma_bf16(a, bn, chn);
  }

  const int nloc = lane & 15;
  const int mhalf = (lane >> 4) * 8;
#pragma unroll
  for (int v = 0; v < 8; ++v) {
    const int m = mt * 16 + mhalf + v;       // batch row
    const int j = jbase + nloc;              // hidden col
    float r  = sigmoidf_(cr[v] + bias[j]);
    float z  = sigmoidf_(cz[v] + bias[Hc + j]);
    float nn = tanhf(cxn[v] + bias[2 * Hc + j] + r * (chn[v] + bias[3 * Hc + j]));
    float hold = h_rd[(size_t)m * Hc + j];
    float hnew = (1.0f - z) * nn + z * hold;
    h_wr[(size_t)m * Hc + j] = hnew;
    unsigned short hb = f2bf(hnew);
    hb_wr[(size_t)m * Hc + j] = hb;
    y[((size_t)m * Tc + t) * Hc + j] = hb;
  }
}

// ---------------- final FC: [B*T,2H] x [2H,O] ----------------
// grid (BT/64, O/32), block 256
__global__ __launch_bounds__(256) void fc_kernel(
    const unsigned short* __restrict__ yf,
    const unsigned short* __restrict__ yb,
    const unsigned short* __restrict__ fcWT,
    const float* __restrict__ fcb,
    float* __restrict__ out) {
  const int tx = threadIdx.x, lane = tx & 31, wave = tx >> 5;
  const int mt = wave & 3, nt = wave >> 2;
  const long rowBase = (long)blockIdx.x * 64 + mt * 16;
  const int  nbase = blockIdx.y * 32 + nt * 16;

  const unsigned short* pa_f = yf + rowBase * Hc;
  const unsigned short* pa_b = yb + rowBase * Hc;
  const unsigned short* pb   = fcWT + (long)nbase * (2 * Hc);

  f32x8 c{};
  // fw half of K
#pragma unroll 2
  for (int kc = 0; kc < Hc; kc += 32) {
    bf16x16 a = load_frag(pa_f + kc, Hc, lane);
    bf16x16 b = load_frag(pb + kc, 2 * Hc, lane);
    __builtin_prefetch(pb + kc + 64, 0, 1);
    c = wmma_bf16(a, b, c);
  }
  // bw half of K
#pragma unroll 2
  for (int kc = 0; kc < Hc; kc += 32) {
    bf16x16 a = load_frag(pa_b + kc, Hc, lane);
    bf16x16 b = load_frag(pb + Hc + kc, 2 * Hc, lane);
    __builtin_prefetch(pb + Hc + kc + 64, 0, 1);
    c = wmma_bf16(a, b, c);
  }

  const int nloc = lane & 15;
  const int mhalf = (lane >> 4) * 8;
#pragma unroll
  for (int v = 0; v < 8; ++v) {
    long row = rowBase + mhalf + v;
    int n = nbase + nloc;
    out[row * Oc + n] = c[v] + fcb[n];
  }
}

// ---------------- host orchestration ----------------
extern "C" void kernel_launch(void* const* d_in, const int* in_sizes, int n_in,
                              void* d_out, int out_size, void* d_ws, size_t ws_size,
                              hipStream_t stream) {
  (void)in_sizes; (void)n_in; (void)out_size; (void)ws_size;
  const float* x = (const float*)d_in[0];
  // order: x, {fw,bw} x {l0,l1} x (Wx, Wh, bx, bh), fc_W, fc_b
  const float* Wx[2][2] = { { (const float*)d_in[1], (const float*)d_in[5] },
                            { (const float*)d_in[9], (const float*)d_in[13] } };
  const float* Wh[2][2] = { { (const float*)d_in[2], (const float*)d_in[6] },
                            { (const float*)d_in[10], (const float*)d_in[14] } };
  const float* bx[2][2] = { { (const float*)d_in[3], (const float*)d_in[7] },
                            { (const float*)d_in[11], (const float*)d_in[15] } };
  const float* bh[2][2] = { { (const float*)d_in[4], (const float*)d_in[8] },
                            { (const float*)d_in[12], (const float*)d_in[16] } };
  const float* fcW = (const float*)d_in[17];
  const float* fcb = (const float*)d_in[18];
  float* out = (float*)d_out;

  // ---- workspace carve-up ----
  char* ws = (char*)d_ws;
  size_t off = 0;
  auto carve = [&](size_t bytes) -> void* {
    void* p = ws + off;
    off += (bytes + 255) & ~(size_t)255;
    return p;
  };
  const int Ktot0 = Dc + Hc;   // 1536
  const int Ktot1 = Hc + Hc;   // 2048
  unsigned short* WT0[2] = { (unsigned short*)carve((size_t)3 * Hc * Ktot0 * 2),
                             (unsigned short*)carve((size_t)3 * Hc * Ktot0 * 2) };
  unsigned short* WT1[2] = { (unsigned short*)carve((size_t)3 * Hc * Ktot1 * 2),
                             (unsigned short*)carve((size_t)3 * Hc * Ktot1 * 2) };
  unsigned short* fcWT = (unsigned short*)carve((size_t)Oc * 2 * Hc * 2);
  float* bias0[2] = { (float*)carve(4 * Hc * 4), (float*)carve(4 * Hc * 4) };
  float* bias1[2] = { (float*)carve(4 * Hc * 4), (float*)carve(4 * Hc * 4) };
  unsigned short* xbf = (unsigned short*)carve((size_t)Bc * Tc * Dc * 2);
  unsigned short* y0[2] = { (unsigned short*)carve((size_t)Bc * Tc * Hc * 2),
                            (unsigned short*)carve((size_t)Bc * Tc * Hc * 2) };
  unsigned short* y1[2] = { (unsigned short*)carve((size_t)Bc * Tc * Hc * 2),
                            (unsigned short*)carve((size_t)Bc * Tc * Hc * 2) };
  float* h32 = (float*)carve((size_t)2 * 2 * BHc * 4);          // [dir][buf][B*H]
  unsigned short* hbf = (unsigned short*)carve((size_t)2 * 2 * BHc * 2);

  // ---- preprocessing ----
  cvt_bf16_kernel<<<2048, 256, 0, stream>>>(x, xbf, (long)Bc * Tc * Dc);
  for (int d = 0; d < 2; ++d) {
    build_wt_kernel<<<2048, 256, 0, stream>>>(Wx[d][0], Wh[d][0], WT0[d], Dc, Ktot0);
    build_wt_kernel<<<2048, 256, 0, stream>>>(Wx[d][1], Wh[d][1], WT1[d], Hc, Ktot1);
    build_bias_kernel<<<16, 256, 0, stream>>>(bx[d][0], bh[d][0], bias0[d]);
    build_bias_kernel<<<16, 256, 0, stream>>>(bx[d][1], bh[d][1], bias1[d]);
  }
  build_fcwt_kernel<<<1024, 256, 0, stream>>>(fcW, fcWT);

  dim3 stepGrid(Hc / 32, 2), stepBlk(256);

  // ---- layer 0 ----
  zero_h_kernel<<<(2 * BHc + 255) / 256, 256, 0, stream>>>(h32, hbf);
  for (int i = 0; i < Tc; ++i) {
    gru_step_kernel<<<stepGrid, stepBlk, 0, stream>>>(
        xbf, xbf, Dc, Ktot0, WT0[0], WT0[1], bias0[0], bias0[1],
        h32, hbf, y0[0], y0[1], i);
  }
  // final hiddens of layer 0 end up in buf 0 (iter 511 writes buf 0)
  hipMemcpyAsync(out + OUT_MAIN + 0 * BHc, h32 + 0 * 2 * BHc,
                 (size_t)BHc * 4, hipMemcpyDeviceToDevice, stream);  // fw_h[0]
  hipMemcpyAsync(out + OUT_MAIN + 2 * BHc, h32 + 1 * 2 * BHc,
                 (size_t)BHc * 4, hipMemcpyDeviceToDevice, stream);  // bw_h[0]

  // ---- layer 1 ----
  zero_h_kernel<<<(2 * BHc + 255) / 256, 256, 0, stream>>>(h32, hbf);
  for (int i = 0; i < Tc; ++i) {
    gru_step_kernel<<<stepGrid, stepBlk, 0, stream>>>(
        y0[0], y0[1], Hc, Ktot1, WT1[0], WT1[1], bias1[0], bias1[1],
        h32, hbf, y1[0], y1[1], i);
  }
  hipMemcpyAsync(out + OUT_MAIN + 1 * BHc, h32 + 0 * 2 * BHc,
                 (size_t)BHc * 4, hipMemcpyDeviceToDevice, stream);  // fw_h[1]
  hipMemcpyAsync(out + OUT_MAIN + 3 * BHc, h32 + 1 * 2 * BHc,
                 (size_t)BHc * 4, hipMemcpyDeviceToDevice, stream);  // bw_h[1]

  // ---- final FC ----
  dim3 fcGrid((Bc * Tc) / 64, Oc / 32);
  fc_kernel<<<fcGrid, 256, 0, stream>>>(y1[0], y1[1], fcWT, fcb, out);
}
